// DeBERTa_4518305595697
// MI455X (gfx1250) — matrix operand order repaired and checked by reading, not compiled, not run on hardware
//
#include <hip/hip_runtime.h>
#include <math.h>

// ---------------- model constants (DeBERTa-base-like, 6 layers) ----------------
#define Bc    4
#define Sc    512
#define Hc    768
#define NHc   12
#define DHc   64
#define FFc   3072
#define Lc    6
#define SPANc 256
#define S2c   (2 * SPANc)   // 512

typedef __bf16 bf16_t;
typedef __attribute__((ext_vector_type(4)))  __bf16 v4bf;
typedef __attribute__((ext_vector_type(8)))  __bf16 v8bf;
typedef __attribute__((ext_vector_type(16))) __bf16 v16bf;
typedef __attribute__((ext_vector_type(8)))  float  v8f;

// =====================================================================
// Generic batched GEMM: C = act(alpha*(A@B + bias)) + Res
//   A: [M,K] f32 row-major (lda), per-z offset zb*sAb + zh*sAh
//   B: [K,N] f32 row-major (ldb)  (transB: B(k,n) = Bm[n*ldb + k])
//   C: [M,N] f32 row-major (ldc)
// Tile: 64 x (16*NT) per 128-thread block (4 waves; wave w owns rows
// w*16..w*16+15, all NT 16-col subtiles). K stepped by 32 with
// double-buffered LDS; math = v_wmma_f32_16x16x32_bf16, f32 accumulate.
// Requires M%64==0, N%(16*NT)==0, K%32==0 (true for all GEMMs here).
// =====================================================================
template <int NT>
__device__ __forceinline__ void stage_tiles(
    const float* __restrict__ A, size_t lda, int m0, int k0,
    const float* __restrict__ Bm, size_t ldb, int n0, int transB,
    bf16_t (*__restrict__ As)[32], bf16_t (*__restrict__ Bt)[32], int t)
{
  // ---- A: 64x32 f32 -> bf16, float4 loads (global_load_b128), b64 LDS stores
  #pragma unroll
  for (int i = 0; i < 4; ++i) {
    const int e = i * 128 + t;                 // 512 float4s total
    const int row = e >> 3, c4 = (e & 7) * 4;  // 8 float4 per 32-wide row
    const float4 f = *(const float4*)&A[(size_t)(m0 + row) * lda + (k0 + c4)];
    v4bf o; o[0] = (bf16_t)f.x; o[1] = (bf16_t)f.y; o[2] = (bf16_t)f.z; o[3] = (bf16_t)f.w;
    *(v4bf*)&As[row][c4] = o;
  }
  // ---- B: 32 x (16*NT) f32 -> Bt[n][k] (K-contiguous per column)
  constexpr int TN   = 16 * NT;
  constexpr int ITER = (32 * TN / 4) / 128;
  if (transB) {  // Bm rows are n (K-contiguous already)
    #pragma unroll
    for (int i = 0; i < ITER; ++i) {
      const int e = i * 128 + t;
      const int n = e >> 3, c4 = (e & 7) * 4;
      const float4 f = *(const float4*)&Bm[(size_t)(n0 + n) * ldb + (k0 + c4)];
      v4bf o; o[0] = (bf16_t)f.x; o[1] = (bf16_t)f.y; o[2] = (bf16_t)f.z; o[3] = (bf16_t)f.w;
      *(v4bf*)&Bt[n][c4] = o;
    }
  } else {       // Bm rows are k; transpose into Bt
    constexpr int F4PR = TN / 4;               // float4 per k-row
    #pragma unroll
    for (int i = 0; i < ITER; ++i) {
      const int e = i * 128 + t;
      const int kk = e / F4PR, nx = (e % F4PR) * 4;
      const float4 f = *(const float4*)&Bm[(size_t)(k0 + kk) * ldb + (n0 + nx)];
      Bt[nx + 0][kk] = (bf16_t)f.x; Bt[nx + 1][kk] = (bf16_t)f.y;
      Bt[nx + 2][kk] = (bf16_t)f.z; Bt[nx + 3][kk] = (bf16_t)f.w;
    }
  }
}

template <int NT>
__global__ __launch_bounds__(128) void gemm_bf16_wmma(
    const float* __restrict__ A, size_t lda, size_t sAb, size_t sAh,
    const float* __restrict__ Bm, size_t ldb, size_t sBb, size_t sBh, int transB,
    float* __restrict__ C, size_t ldc, size_t sCb, size_t sCh,
    const float* __restrict__ bias,
    const float* __restrict__ Res, size_t ldr,
    int M, int N, int K, float alpha, int act, int nhdiv)
{
  const int z  = blockIdx.z;
  const int zb = z / nhdiv, zh = z % nhdiv;
  A  += (size_t)zb * sAb + (size_t)zh * sAh;
  Bm += (size_t)zb * sBb + (size_t)zh * sBh;
  C  += (size_t)zb * sCb + (size_t)zh * sCh;

  const int m0 = blockIdx.y * 64;
  const int n0 = blockIdx.x * (16 * NT);
  const int t     = threadIdx.x;
  const int wave  = t >> 5;
  const int lane  = t & 31;
  const int lhalf = lane >> 4;   // 0: lanes 0-15, 1: lanes 16-31
  const int l16   = lane & 15;

  __shared__ __align__(64) bf16_t As[2][64][32];        // [buf][m][k]
  __shared__ __align__(64) bf16_t Bt[2][16 * NT][32];   // [buf][n][k]

  v8f acc[NT] = {};

  const int nK = K >> 5;
  stage_tiles<NT>(A, lda, m0, 0, Bm, ldb, n0, transB, As[0], Bt[0], t);

  for (int kt = 0; kt < nK; ++kt) {
    __syncthreads();                       // buf[kt&1] staged; buf[1-kt&1] free
    const int cur = kt & 1;
    if (kt + 1 < nK) {
      stage_tiles<NT>(A, lda, m0, (kt + 1) * 32, Bm, ldb, n0, transB,
                      As[cur ^ 1], Bt[cur ^ 1], t);
      if (kt + 2 < nK)  // global_prefetch_b8 for the K-tile after next
        __builtin_prefetch((const void*)&A[(size_t)(m0 + (t >> 1)) * lda + (kt + 2) * 32], 0, 0);
    }

    // ---- A fragment (16x32 bf16): lane = row; half picks K phase ----
    // lanes 0-15: K = 0..15 ; lanes 16-31: K = 16..31
    const int arow = wave * 16 + l16;
    const int kb   = lhalf * 16;
    const v8bf alo = *(const v8bf*)&As[cur][arow][kb];
    const v8bf ahi = *(const v8bf*)&As[cur][arow][kb + 8];
    v16bf af;
    #pragma unroll
    for (int i = 0; i < 8; ++i) { af[i] = alo[i]; af[i + 8] = ahi[i]; }

    // ---- B fragments in groups of 4, then uninterrupted WMMA chain ----
    #pragma unroll
    for (int g = 0; g < NT; g += 4) {
      v16bf bf[4];
      #pragma unroll
      for (int j = 0; j < 4; ++j)
        bf[j] = *(const v16bf*)&Bt[cur][(g + j) * 16 + l16][lhalf * 16];
      #pragma unroll
      for (int j = 0; j < 4; ++j)
        acc[g + j] = __builtin_amdgcn_wmma_f32_16x16x32_bf16(
            false, af, false, bf[j], (short)0, acc[g + j], false, false);
    }
  }

  // ---- epilogue: C/D layout M = r + 8*half, N = lane%16 ----
  #pragma unroll
  for (int nt = 0; nt < NT; ++nt) {
    #pragma unroll
    for (int r = 0; r < 8; ++r) {
      const int m = m0 + wave * 16 + r + 8 * lhalf;
      const int n = n0 + nt * 16 + l16;
      float v = acc[nt][r];
      if (bias) v += bias[n];
      v *= alpha;
      if (act == 1) v = 0.5f * v * (1.0f + erff(v * 0.70710678118654752440f)); // exact gelu
      if (Res) v += Res[(size_t)m * ldr + n];
      C[(size_t)m * ldc + n] = v;
    }
  }
}

// =====================================================================
// block reductions (blockDim.x == 256)
// =====================================================================
__device__ __forceinline__ float blk_sum(float v, float* sm) {
  const int t = threadIdx.x;
  sm[t] = v; __syncthreads();
  for (int s = 128; s > 0; s >>= 1) { if (t < s) sm[t] += sm[t + s]; __syncthreads(); }
  float r = sm[0]; __syncthreads(); return r;
}
__device__ __forceinline__ float blk_max(float v, float* sm) {
  const int t = threadIdx.x;
  sm[t] = v; __syncthreads();
  for (int s = 128; s > 0; s >>= 1) { if (t < s) sm[t] = fmaxf(sm[t], sm[t + s]); __syncthreads(); }
  float r = sm[0]; __syncthreads(); return r;
}

// =====================================================================
// Embeddings + MaskedLayerNorm: one block per token, 256 thr x 3 elems
// =====================================================================
__global__ __launch_bounds__(256) void embed_ln_kernel(
    const int* __restrict__ ids, const int* __restrict__ tts, const int* __restrict__ am,
    const float* __restrict__ wemb, const float* __restrict__ temb,
    const float* __restrict__ g, const float* __restrict__ bta, float* __restrict__ out)
{
  __shared__ float sm[256];
  const int tok = blockIdx.x;
  const int t = threadIdx.x;
  const float* we = wemb + (size_t)ids[tok] * Hc;
  const float* te = temb + (size_t)tts[tok] * Hc;
  float x[3]; float s = 0.f;
  #pragma unroll
  for (int i = 0; i < 3; ++i) { int idx = t + i * 256; x[i] = we[idx] + te[idx]; s += x[i]; }
  const float mean = blk_sum(s, sm) * (1.0f / Hc);
  float s2 = 0.f;
  #pragma unroll
  for (int i = 0; i < 3; ++i) { float d = x[i] - mean; s2 += d * d; }
  const float var  = blk_sum(s2, sm) * (1.0f / Hc);
  const float rstd = rsqrtf(var + 1e-7f);
  const float mk   = (float)am[tok];
  float* o = out + (size_t)tok * Hc;
  #pragma unroll
  for (int i = 0; i < 3; ++i) {
    int idx = t + i * 256;
    o[idx] = ((x[i] - mean) * rstd * g[idx] + bta[idx]) * mk;
  }
}

// =====================================================================
// LayerNorm: one block per row of [*, H]
// =====================================================================
__global__ __launch_bounds__(256) void ln_kernel(
    const float* __restrict__ X, const float* __restrict__ g,
    const float* __restrict__ bta, float* __restrict__ out)
{
  __shared__ float sm[256];
  const int row = blockIdx.x;
  const int t = threadIdx.x;
  const float* x = X + (size_t)row * Hc;
  float v[3]; float s = 0.f;
  #pragma unroll
  for (int i = 0; i < 3; ++i) { int idx = t + i * 256; v[i] = x[idx]; s += v[i]; }
  const float mean = blk_sum(s, sm) * (1.0f / Hc);
  float s2 = 0.f;
  #pragma unroll
  for (int i = 0; i < 3; ++i) { float d = v[i] - mean; s2 += d * d; }
  const float var  = blk_sum(s2, sm) * (1.0f / Hc);
  const float rstd = rsqrtf(var + 1e-7f);
  float* o = out + (size_t)row * Hc;
  #pragma unroll
  for (int i = 0; i < 3; ++i) {
    int idx = t + i * 256;
    o[idx] = (v[i] - mean) * rstd * g[idx] + bta[idx];
  }
}

// =====================================================================
// score = qk + gather(c2p, r) + gather(p2c, r) ; XSoftmax -> probs in place
// Both gathers use r = clip(q-k+SPAN, 0, 2*SPAN-1):
//   c2p term: c2p[b,h,q,r] ; p2c term: p2c[b,h,k,r] (post-transpose form)
// =====================================================================
__global__ __launch_bounds__(256) void score_softmax_kernel(
    float* __restrict__ qk, const float* __restrict__ c2p,
    const float* __restrict__ p2c, const int* __restrict__ am)
{
  __shared__ float sm[256];
  const int q = blockIdx.x, h = blockIdx.y, b = blockIdx.z;
  const int t = threadIdx.x;
  const size_t rowbase = (((size_t)b * NHc + h) * Sc + q) * Sc;
  const size_t c2pbase = (((size_t)b * NHc + h) * Sc + q) * S2c;
  const size_t p2cslab = ((size_t)b * NHc + h) * (size_t)Sc * S2c;
  const int qv = am[b * Sc + q];
  const float NEG = -3.4028234663852886e38f;  // finfo(f32).min
  float sv[2]; int valid[2];
  float mx = NEG;
  #pragma unroll
  for (int i = 0; i < 2; ++i) {
    const int k = t + i * 256;
    int r = q - k + SPANc; r = r < 0 ? 0 : (r > S2c - 1 ? S2c - 1 : r);
    float s = qk[rowbase + k] + c2p[c2pbase + r] + p2c[p2cslab + (size_t)k * S2c + r];
    valid[i] = (qv != 0) && (am[b * Sc + k] != 0);
    sv[i] = valid[i] ? s : NEG;
    mx = fmaxf(mx, sv[i]);
  }
  mx = blk_max(mx, sm);
  float es = 0.f;
  #pragma unroll
  for (int i = 0; i < 2; ++i) { sv[i] = __expf(sv[i] - mx); es += sv[i]; }
  es = blk_sum(es, sm);
  const float inv = 1.0f / es;
  #pragma unroll
  for (int i = 0; i < 2; ++i) {
    const int k = t + i * 256;
    qk[rowbase + k] = valid[i] ? sv[i] * inv : 0.0f;   // XSoftmax re-zero
  }
}

// =====================================================================
// host orchestration
// =====================================================================
extern "C" void kernel_launch(void* const* d_in, const int* in_sizes, int n_in,
                              void* d_out, int out_size, void* d_ws, size_t ws_size,
                              hipStream_t stream)
{
  const int*   input_ids  = (const int*)d_in[0];
  const int*   token_type = (const int*)d_in[1];
  const int*   amask      = (const int*)d_in[2];
  const float* word_emb   = (const float*)d_in[3];
  const float* tok_emb    = (const float*)d_in[4];
  const float* rel_emb    = (const float*)d_in[5];
  const float* eg  = (const float*)d_in[6];
  const float* ebi = (const float*)d_in[7];
  const float* Wq  = (const float*)d_in[8];
  const float* bq  = (const float*)d_in[9];
  const float* Wk  = (const float*)d_in[10];
  const float* Wv  = (const float*)d_in[11];
  const float* bv  = (const float*)d_in[12];
  const float* Wpk = (const float*)d_in[13];
  const float* Wpq = (const float*)d_in[14];
  const float* bpq = (const float*)d_in[15];
  const float* Wo  = (const float*)d_in[16];
  const float* bo  = (const float*)d_in[17];
  const float* l1g = (const float*)d_in[18];
  const float* l1b = (const float*)d_in[19];
  const float* Wi  = (const float*)d_in[20];
  const float* bi  = (const float*)d_in[21];
  const float* Wo2 = (const float*)d_in[22];
  const float* bo2 = (const float*)d_in[23];
  const float* l2g = (const float*)d_in[24];
  const float* l2b = (const float*)d_in[25];
  float* out = (float*)d_out;
  (void)in_sizes; (void)n_in; (void)out_size; (void)ws_size;

  const size_t MT = (size_t)Bc * Sc;     // 2048 tokens
  const size_t HS = MT * Hc;
  float* p = (float*)d_ws;
  auto take = [&](size_t n) { float* r = p; p += n; return r; };
  float* hA    = take(HS);
  float* hB    = take(HS);
  float* tmp   = take(HS);
  float* Q     = take(HS);
  float* Kb    = take(HS);
  float* V     = take(HS);
  float* ctx   = take(HS);
  float* inter = take(MT * FFc);
  float* posk  = take((size_t)S2c * Hc);
  float* posq  = take((size_t)S2c * Hc);
  const size_t SS = (size_t)Bc * NHc * Sc * Sc;
  float* qk    = take(SS);
  float* c2p   = take(SS);
  float* p2c   = take(SS);

  const float invScale = 1.0f / 13.856406460551018f;  // 1/sqrt(DH*3)

  auto gemm = [&](const float* A, size_t lda, size_t sAb, size_t sAh,
                  const float* Bm, size_t ldb, size_t sBb, size_t sBh, int trB,
                  float* C, size_t ldc, size_t sCb, size_t sCh,
                  const float* bias, const float* Res, size_t ldr,
                  int M, int N, int K, float alpha, int act, int nz, int nhdiv) {
    if (N % 128 == 0) {
      dim3 grid((unsigned)(N / 128), (unsigned)(M / 64), (unsigned)nz);
      gemm_bf16_wmma<8><<<grid, 128, 0, stream>>>(A, lda, sAb, sAh, Bm, ldb, sBb, sBh, trB,
                                                  C, ldc, sCb, sCh, bias, Res, ldr,
                                                  M, N, K, alpha, act, nhdiv);
    } else {
      dim3 grid((unsigned)(N / 64), (unsigned)(M / 64), (unsigned)nz);
      gemm_bf16_wmma<4><<<grid, 128, 0, stream>>>(A, lda, sAb, sAh, Bm, ldb, sBb, sBh, trB,
                                                  C, ldc, sCb, sCh, bias, Res, ldr,
                                                  M, N, K, alpha, act, nhdiv);
    }
  };

  embed_ln_kernel<<<(unsigned)MT, 256, 0, stream>>>(input_ids, token_type, amask,
                                                    word_emb, tok_emb, eg, ebi, hA);

  const size_t sBH = (size_t)Sc * Hc;
  for (int l = 0; l < Lc; ++l) {
    const size_t wHH = (size_t)l * Hc * Hc;
    const size_t wHF = (size_t)l * Hc * FFc;
    // --- QKV projections (q pre-scaled by 1/sqrt(3*DH), k has no bias) ---
    gemm(hA, Hc,0,0, Wq + wHH, Hc,0,0,0, Q,  Hc,0,0, bq + (size_t)l*Hc, nullptr,0,
         (int)MT, Hc, Hc, invScale, 0, 1, 1);
    gemm(hA, Hc,0,0, Wk + wHH, Hc,0,0,0, Kb, Hc,0,0, nullptr,           nullptr,0,
         (int)MT, Hc, Hc, 1.0f,     0, 1, 1);
    gemm(hA, Hc,0,0, Wv + wHH, Hc,0,0,0, V,  Hc,0,0, bv + (size_t)l*Hc, nullptr,0,
         (int)MT, Hc, Hc, 1.0f,     0, 1, 1);
    // --- relative-position projections: pos_k, pos_q (scaled) [2*SPAN, H] ---
    gemm(rel_emb, Hc,0,0, Wpk + wHH, Hc,0,0,0, posk, Hc,0,0, nullptr, nullptr,0,
         S2c, Hc, Hc, 1.0f,     0, 1, 1);
    gemm(rel_emb, Hc,0,0, Wpq + wHH, Hc,0,0,0, posq, Hc,0,0, bpq + (size_t)l*Hc, nullptr,0,
         S2c, Hc, Hc, invScale, 0, 1, 1);
    // --- per-(b,h) score GEMMs (z = b*NH + h, slices strided by DH) ---
    gemm(Q,  Hc, sBH, DHc, Kb,   Hc, sBH, DHc, 1, qk,  Sc,  (size_t)NHc*Sc*Sc,  (size_t)Sc*Sc,
         nullptr, nullptr,0, Sc, Sc,  DHc, 1.0f, 0, Bc*NHc, NHc);
    gemm(Q,  Hc, sBH, DHc, posk, Hc, 0,   DHc, 1, c2p, S2c, (size_t)NHc*Sc*S2c, (size_t)Sc*S2c,
         nullptr, nullptr,0, Sc, S2c, DHc, 1.0f, 0, Bc*NHc, NHc);
    gemm(Kb, Hc, sBH, DHc, posq, Hc, 0,   DHc, 1, p2c, S2c, (size_t)NHc*Sc*S2c, (size_t)Sc*S2c,
         nullptr, nullptr,0, Sc, S2c, DHc, 1.0f, 0, Bc*NHc, NHc);
    // --- combine + masked softmax (probs written back into qk) ---
    score_softmax_kernel<<<dim3(Sc, NHc, Bc), 256, 0, stream>>>(qk, c2p, p2c, amask);
    // --- context: probs @ V -> ctx [B,S,H]  (N=64 -> NT=4 path) ---
    gemm(qk, Sc, (size_t)NHc*Sc*Sc, (size_t)Sc*Sc, V, Hc, sBH, DHc, 0,
         ctx, Hc, sBH, DHc, nullptr, nullptr,0, Sc, DHc, Sc, 1.0f, 0, Bc*NHc, NHc);
    // --- output projection + residual, LN1 ---
    gemm(ctx, Hc,0,0, Wo + wHH, Hc,0,0,0, tmp, Hc,0,0, bo + (size_t)l*Hc, hA, Hc,
         (int)MT, Hc, Hc, 1.0f, 0, 1, 1);
    ln_kernel<<<(unsigned)MT, 256, 0, stream>>>(tmp, l1g + (size_t)l*Hc, l1b + (size_t)l*Hc, hB);
    // --- FFN: gelu(hB@Wi + bi) @ Wo2 + bo2 + hB, LN2 ---
    gemm(hB, Hc,0,0, Wi + wHF, FFc,0,0,0, inter, FFc,0,0, bi + (size_t)l*FFc, nullptr,0,
         (int)MT, FFc, Hc, 1.0f, 1, 1, 1);
    gemm(inter, FFc,0,0, Wo2 + wHF, Hc,0,0,0, tmp, Hc,0,0, bo2 + (size_t)l*Hc, hB, Hc,
         (int)MT, Hc, FFc, 1.0f, 0, 1, 1);
    ln_kernel<<<(unsigned)MT, 256, 0, stream>>>(tmp, l2g + (size_t)l*Hc, l2b + (size_t)l*Hc,
                                                (l == Lc - 1) ? out : hA);
  }
}
